// Encoder_64201171141373
// MI455X (gfx1250) — matrix operand order repaired and checked by reading, not compiled, not run on hardware
//
#include <hip/hip_runtime.h>
#include <hip/hip_bf16.h>
#include <math.h>

// ---------------------------------------------------------------------------
// Transformer encoder forward for gfx1250 (MI455X), wave32 + WMMA f16 +
// 4-deep async global->LDS pipeline (ASYNCcnt counted waits).
// B=16 S=512 D=1024 H=16 HD=64 F=4096 L=6
// ---------------------------------------------------------------------------

typedef __attribute__((ext_vector_type(16))) _Float16 v16h;
typedef __attribute__((ext_vector_type(8)))  _Float16 v8h;
typedef __attribute__((ext_vector_type(4)))  _Float16 v4h;
typedef __attribute__((ext_vector_type(2)))  _Float16 v2h;
typedef __attribute__((ext_vector_type(8)))  float    v8f;

#define Bc 16
#define Sc 512
#define Dc 1024
#define Hc 16
#define HDc 64
#define Fc 4096
#define Lc 6
#define Mrows (Bc*Sc)
#define FFN_CHUNK 2048

__device__ __forceinline__ v8f wmma_f16(v16h a, v16h b, v8f c) {
  return __builtin_amdgcn_wmma_f32_16x16x32_f16(
      false, a, false, b, (short)0, c, false, false);
}

__device__ __forceinline__ v16h combine16(v8h lo, v8h hi) {
  v16h r;
#pragma unroll
  for (int i = 0; i < 8; ++i) { r[i] = lo[i]; r[i + 8] = hi[i]; }
  return r;
}

// Async 16-byte global->LDS copy (ASYNCcnt), saddr form.
__device__ __forceinline__ void async_ld_b128(unsigned lds_off, unsigned goff,
                                              const void* base) {
  asm volatile("global_load_async_to_lds_b128 %0, %1, %2"
               :: "v"(lds_off), "v"(goff), "s"(base) : "memory");
}
template <int N>
__device__ __forceinline__ void wait_async() {
  asm volatile("s_wait_asynccnt %0" :: "i"(N) : "memory");
}

// ---------------------------------------------------------------------------
// Embedding + sinusoidal positional encoding (writes fp32 residual stream)
// ---------------------------------------------------------------------------
__global__ __launch_bounds__(256) void embed_kernel(
    const int* __restrict__ tokens, const float* __restrict__ emb,
    float* __restrict__ X) {
  const int bs  = blockIdx.x;
  const int s   = bs & (Sc - 1);
  const int tok = tokens[bs];
  const int d0  = threadIdx.x * 4;
  const float kLog = -logf(10000.0f) / (float)Dc;
  float4 e4 = *(const float4*)(emb + (size_t)tok * Dc + d0);
  float a0 = (float)s * expf((float)(d0)     * kLog);
  float a1 = (float)s * expf((float)(d0 + 2) * kLog);
  float4 o;
  o.x = e4.x * 32.0f + sinf(a0);
  o.y = e4.y * 32.0f + cosf(a0);
  o.z = e4.z * 32.0f + sinf(a1);
  o.w = e4.w * 32.0f + cosf(a1);
  *(float4*)(X + (size_t)bs * Dc + d0) = o;
}

// ---------------------------------------------------------------------------
// LayerNorm over D=1024; fp32 in, optional fp32 and/or f16 out.
// ---------------------------------------------------------------------------
__global__ __launch_bounds__(256) void ln_kernel(
    const float* __restrict__ X, const float* __restrict__ gam,
    const float* __restrict__ bet, float* Yf, _Float16* Yh) {
  __shared__ float red_s[8];
  __shared__ float red_ss[8];
  const int tid  = threadIdx.x;
  const int wid  = tid >> 5;
  const int lane = tid & 31;
  float4 v = *(const float4*)(X + (size_t)blockIdx.x * Dc + tid * 4);
  float s  = v.x + v.y + v.z + v.w;
  float ss = v.x * v.x + v.y * v.y + v.z * v.z + v.w * v.w;
#pragma unroll
  for (int m = 1; m <= 16; m <<= 1) {
    s  += __shfl_xor(s,  m, 32);
    ss += __shfl_xor(ss, m, 32);
  }
  if (lane == 0) { red_s[wid] = s; red_ss[wid] = ss; }
  __syncthreads();
  float tot = 0.f, tots = 0.f;
#pragma unroll
  for (int w = 0; w < 8; ++w) { tot += red_s[w]; tots += red_ss[w]; }
  const float mean = tot * (1.0f / Dc);
  const float var  = tots * (1.0f / Dc) - mean * mean;
  const float rstd = rsqrtf(var + 1e-5f);
  float4 g4 = *(const float4*)(gam + tid * 4);
  float4 b4 = *(const float4*)(bet + tid * 4);
  float4 o;
  o.x = (v.x - mean) * rstd * g4.x + b4.x;
  o.y = (v.y - mean) * rstd * g4.y + b4.y;
  o.z = (v.z - mean) * rstd * g4.z + b4.z;
  o.w = (v.w - mean) * rstd * g4.w + b4.w;
  if (Yf) *(float4*)(Yf + (size_t)blockIdx.x * Dc + tid * 4) = o;
  if (Yh) {
    v4h h;
    h[0] = (_Float16)o.x; h[1] = (_Float16)o.y;
    h[2] = (_Float16)o.z; h[3] = (_Float16)o.w;
    *(v4h*)(Yh + (size_t)blockIdx.x * Dc + tid * 4) = h;
  }
}

// ---------------------------------------------------------------------------
// Weight transpose+convert: W[K][N] f32 -> WT[N][K] f16 (LDS-tiled 32x32)
// ---------------------------------------------------------------------------
__global__ __launch_bounds__(256) void wtrans_kernel(
    const float* __restrict__ W, _Float16* __restrict__ WT, int K, int N) {
  __shared__ float T[32][33];
  const int n0 = blockIdx.x * 32;
  const int k0 = blockIdx.y * 32;
  const int tid = threadIdx.x;
#pragma unroll
  for (int ii = 0; ii < 4; ++ii) {
    int idx = tid + ii * 256;
    int kr = idx >> 5, nc = idx & 31;
    T[kr][nc] = W[(size_t)(k0 + kr) * N + n0 + nc];
  }
  __syncthreads();
#pragma unroll
  for (int ii = 0; ii < 2; ++ii) {
    int pid = tid + ii * 256;
    int nr = pid >> 4, kc2 = (pid & 15) * 2;
    v2h h;
    h[0] = (_Float16)T[kc2][nr];
    h[1] = (_Float16)T[kc2 + 1][nr];
    *(v2h*)&WT[(size_t)(n0 + nr) * K + k0 + kc2] = h;
  }
}

// ---------------------------------------------------------------------------
// WMMA GEMM:  C[M,N] = act( A16[M,K] @ WT16[N,K]^T + bias[N] ) (+ residual)
// 8 waves, 64(M)x128(N) block tile, K-step 32, 4-deep async pipeline.
// Per K-step per wave: 2 A-frags x 2 B-frags -> 4 WMMA.
// Stage = 3 async b128/thread (1 A + 2 B); s_wait_asynccnt 6 drains oldest.
// A frag: halves e -> K = e+8g / e+16+8g ; B frag: halves e -> K = e+8g / e+16+8g
// ---------------------------------------------------------------------------
__global__ __launch_bounds__(256) void gemm_kernel(
    const _Float16* __restrict__ A, const _Float16* __restrict__ WT,
    const float* __restrict__ bias, const float* R, float* Cf, _Float16* Ch,
    int M, int N, int K, int relu) {
  __shared__ __align__(16) _Float16 As[4][64][40];
  __shared__ __align__(16) _Float16 Bs[4][128][40];
  const int tid  = threadIdx.x;
  const int bM   = blockIdx.y * 64;
  const int bN   = blockIdx.x * 128;
  const int wid  = tid >> 5;
  const int lane = tid & 31;
  const int r    = lane & 15;
  const int g    = lane >> 4;
  const int wm   = wid >> 2;        // 0..1 -> 32 rows
  const int wn   = wid & 3;         // 0..3 -> 32 cols
  const int miA  = tid >> 2;        // 0..63
  const int ki8  = (tid & 3) * 8;   // 0,8,16,24
  const int KT   = K >> 5;
  v8f acc00 = {}, acc01 = {}, acc10 = {}, acc11 = {};

  auto stage = [&](int t) {
    const int k0  = (t < KT ? t : KT - 1) << 5;   // clamped tail overfetch
    const int buf = t & 3;
    async_ld_b128((unsigned)(size_t)&As[buf][miA][ki8],
                  (unsigned)(((size_t)(bM + miA) * K + k0 + ki8) * 2), A);
#pragma unroll
    for (int ii = 0; ii < 2; ++ii) {
      int cid = tid + ii * 256;                   // < 512 chunks
      int ni = cid >> 2, kb8 = (cid & 3) * 8;
      async_ld_b128((unsigned)(size_t)&Bs[buf][ni][kb8],
                    (unsigned)(((size_t)(bN + ni) * K + k0 + kb8) * 2), WT);
    }
  };

  stage(0); stage(1); stage(2);                   // 9 asyncs in flight

  for (int kt = 0; kt < KT; ++kt) {
    wait_async<6>();                              // oldest stage (tile kt) done
    __syncthreads();
    stage(kt + 3);                                // into buffer of tile kt-1
    const int buf = kt & 3;
    const int am0 = wm * 32 + r;
    const int bn0 = wn * 32 + r;
    v16h av0 = combine16(*(const v8h*)&As[buf][am0][8 * g],
                         *(const v8h*)&As[buf][am0][16 + 8 * g]);
    v16h av1 = combine16(*(const v8h*)&As[buf][am0 + 16][8 * g],
                         *(const v8h*)&As[buf][am0 + 16][16 + 8 * g]);
    v16h bv0 = combine16(*(const v8h*)&Bs[buf][bn0][8 * g],
                         *(const v8h*)&Bs[buf][bn0][16 + 8 * g]);
    v16h bv1 = combine16(*(const v8h*)&Bs[buf][bn0 + 16][8 * g],
                         *(const v8h*)&Bs[buf][bn0 + 16][16 + 8 * g]);
    acc00 = wmma_f16(av0, bv0, acc00);
    acc10 = wmma_f16(av1, bv0, acc10);
    acc01 = wmma_f16(av0, bv1, acc01);
    acc11 = wmma_f16(av1, bv1, acc11);
  }

#pragma unroll
  for (int t = 0; t < 2; ++t) {
    const int col = bN + wn * 32 + t * 16 + r;
    const float bb = bias[col];
#pragma unroll
    for (int s = 0; s < 2; ++s) {
      const int row0 = bM + wm * 32 + s * 16 + 8 * g;
      const v8f& acc = (s == 0) ? (t == 0 ? acc00 : acc01)
                                : (t == 0 ? acc10 : acc11);
#pragma unroll
      for (int i = 0; i < 8; ++i) {
        float vv = acc[i] + bb;
        if (relu) vv = fmaxf(vv, 0.0f);
        size_t o = (size_t)(row0 + i) * N + col;
        if (R) vv += R[o];
        if (Cf) Cf[o] = vv;
        if (Ch) Ch[o] = (_Float16)vv;
      }
    }
  }
}

// ---------------------------------------------------------------------------
// Flash attention (f16 in / f16 out): block = 128 thr (4 waves), one (b,h)
// per block; K tile staged via async b128, V tile staged transposed; one wave
// per 16-row q tile; online softmax; P via per-wave LDS relayout.
// ---------------------------------------------------------------------------
__global__ __launch_bounds__(128) void attn_kernel(
    const _Float16* __restrict__ Q, const _Float16* __restrict__ Kt,
    const _Float16* __restrict__ V, _Float16* __restrict__ O) {
  __shared__ __align__(16) _Float16 Ks[32][72];     // [key][d]
  __shared__ __align__(16) _Float16 Vs[64][40];     // [d][key] transposed
  __shared__ __align__(16) _Float16 P[4][16 * 32];  // per-wave P buffer
  const int tid  = threadIdx.x;
  const int wid  = tid >> 5;
  const int lane = tid & 31;
  const int r    = lane & 15;
  const int g    = lane >> 4;
  const int bh   = blockIdx.x >> 3;
  const int qblk = blockIdx.x & 7;
  const int b    = bh >> 4;
  const int hh   = bh & 15;
  const int qt   = qblk * 4 + wid;
  const int qrow = qt * 16 + r;

  const _Float16* qp = Q + ((size_t)(b * Sc + qrow)) * Dc + hh * HDc;
  const v16h a0 = combine16(*(const v8h*)(qp + 8 * g),
                            *(const v8h*)(qp + 16 + 8 * g));
  const v16h a1 = combine16(*(const v8h*)(qp + 32 + 8 * g),
                            *(const v8h*)(qp + 48 + 8 * g));

  const _Float16* kbase = Kt + ((size_t)(b * Sc)) * Dc + hh * HDc;
  const _Float16* vbase = V  + ((size_t)(b * Sc)) * Dc + hh * HDc;

  float m[8], l[8];
#pragma unroll
  for (int i = 0; i < 8; ++i) { m[i] = -1e30f; l[i] = 0.0f; }
  v8f o0 = {}, o1 = {}, o2 = {}, o3 = {};

  for (int kb = 0; kb < Sc; kb += 32) {
    __syncthreads();
#pragma unroll
    for (int ii = 0; ii < 2; ++ii) {
      int cid = tid + ii * 128;
      int key = cid >> 2, d8 = (cid & 3) * 8;
      async_ld_b128((unsigned)(size_t)&Ks[key][d8],
                    (unsigned)(((size_t)(kb + key) * Dc + d8) * 2), kbase);
    }
#pragma unroll
    for (int ii = 0; ii < 4; ++ii) {
      int gid = tid + ii * 128;
      int d = gid & 63, k4 = (gid >> 6) * 4;
      const _Float16* vp = vbase + (size_t)(kb + k4) * Dc + d;
      v4h h;
      h[0] = vp[0];
      h[1] = vp[(size_t)Dc];
      h[2] = vp[2 * (size_t)Dc];
      h[3] = vp[3 * (size_t)Dc];
      *(v4h*)&Vs[d][k4] = h;
    }
    wait_async<0>();
    __syncthreads();

    v16h b00 = combine16(*(const v8h*)&Ks[r][8 * g],
                         *(const v8h*)&Ks[r][16 + 8 * g]);
    v16h b01 = combine16(*(const v8h*)&Ks[r][32 + 8 * g],
                         *(const v8h*)&Ks[r][48 + 8 * g]);
    v16h b10 = combine16(*(const v8h*)&Ks[16 + r][8 * g],
                         *(const v8h*)&Ks[16 + r][16 + 8 * g]);
    v16h b11 = combine16(*(const v8h*)&Ks[16 + r][32 + 8 * g],
                         *(const v8h*)&Ks[16 + r][48 + 8 * g]);
    v8f s0 = {}, s1 = {};
    s0 = wmma_f16(a0, b00, s0); s0 = wmma_f16(a1, b01, s0);
    s1 = wmma_f16(a0, b10, s1); s1 = wmma_f16(a1, b11, s1);

#pragma unroll
    for (int i = 0; i < 8; ++i) {
      float x0 = s0[i] * 0.125f;
      float x1 = s1[i] * 0.125f;
      float mx = fmaxf(x0, x1);
      mx = fmaxf(mx, __shfl_xor(mx, 1, 32));
      mx = fmaxf(mx, __shfl_xor(mx, 2, 32));
      mx = fmaxf(mx, __shfl_xor(mx, 4, 32));
      mx = fmaxf(mx, __shfl_xor(mx, 8, 32));
      float nm = fmaxf(m[i], mx);
      float sc = expf(m[i] - nm);
      float p0 = expf(x0 - nm);
      float p1 = expf(x1 - nm);
      float rs = p0 + p1;
      rs += __shfl_xor(rs, 1, 32);
      rs += __shfl_xor(rs, 2, 32);
      rs += __shfl_xor(rs, 4, 32);
      rs += __shfl_xor(rs, 8, 32);
      l[i] = l[i] * sc + rs;
      m[i] = nm;
      o0[i] *= sc; o1[i] *= sc; o2[i] *= sc; o3[i] *= sc;
      P[wid][(8 * g + i) * 32 + r]      = (_Float16)p0;
      P[wid][(8 * g + i) * 32 + 16 + r] = (_Float16)p1;
    }
    asm volatile("s_wait_dscnt 0" ::: "memory");

    v16h pa = combine16(*(const v8h*)&P[wid][r * 32 + 8 * g],
                        *(const v8h*)&P[wid][r * 32 + 16 + 8 * g]);
    v16h bv0 = combine16(*(const v8h*)&Vs[r][8 * g],
                         *(const v8h*)&Vs[r][16 + 8 * g]);
    v16h bv1 = combine16(*(const v8h*)&Vs[16 + r][8 * g],
                         *(const v8h*)&Vs[16 + r][16 + 8 * g]);
    v16h bv2 = combine16(*(const v8h*)&Vs[32 + r][8 * g],
                         *(const v8h*)&Vs[32 + r][16 + 8 * g]);
    v16h bv3 = combine16(*(const v8h*)&Vs[48 + r][8 * g],
                         *(const v8h*)&Vs[48 + r][16 + 8 * g]);
    o0 = wmma_f16(pa, bv0, o0);
    o1 = wmma_f16(pa, bv1, o1);
    o2 = wmma_f16(pa, bv2, o2);
    o3 = wmma_f16(pa, bv3, o3);
  }

#pragma unroll
  for (int i = 0; i < 8; ++i) {
    float inv = 1.0f / l[i];
    int row = qt * 16 + 8 * g + i;
    _Float16* op = O + ((size_t)(b * Sc + row)) * Dc + hh * HDc + r;
    op[0]  = (_Float16)(o0[i] * inv);
    op[16] = (_Float16)(o1[i] * inv);
    op[32] = (_Float16)(o2[i] * inv);
    op[48] = (_Float16)(o3[i] * inv);
  }
}

// ---------------------------------------------------------------------------
// Host-side orchestration
// ---------------------------------------------------------------------------
extern "C" void kernel_launch(void* const* d_in, const int* in_sizes, int n_in,
                              void* d_out, int out_size, void* d_ws, size_t ws_size,
                              hipStream_t stream) {
  const int*   tokens = (const int*)  d_in[0];
  const float* emb    = (const float*)d_in[1];
  const float* ln1_g  = (const float*)d_in[2];
  const float* ln1_b  = (const float*)d_in[3];
  const float* Wq     = (const float*)d_in[4];
  const float* bq     = (const float*)d_in[5];
  const float* Wk     = (const float*)d_in[6];
  const float* bk     = (const float*)d_in[7];
  const float* Wv     = (const float*)d_in[8];
  const float* bv     = (const float*)d_in[9];
  const float* Wo     = (const float*)d_in[10];
  const float* bo     = (const float*)d_in[11];
  const float* ln2_g  = (const float*)d_in[12];
  const float* ln2_b  = (const float*)d_in[13];
  const float* W1     = (const float*)d_in[14];
  const float* b1     = (const float*)d_in[15];
  const float* W2     = (const float*)d_in[16];
  const float* b2     = (const float*)d_in[17];
  const float* lnf_g  = (const float*)d_in[18];
  const float* lnf_b  = (const float*)d_in[19];

  const size_t ND = (size_t)Mrows * Dc;
  char* w = (char*)d_ws;
  float*     x    = (float*)w;       w += ND * 4;
  _Float16*  h16  = (_Float16*)w;    w += ND * 2;
  _Float16*  q16  = (_Float16*)w;    w += ND * 2;
  _Float16*  k16  = (_Float16*)w;    w += ND * 2;
  _Float16*  v16  = (_Float16*)w;    w += ND * 2;
  _Float16*  f16b = (_Float16*)w;    w += (size_t)FFN_CHUNK * Fc * 2;
  _Float16*  wtA  = (_Float16*)w;    w += (size_t)Dc * Fc * 2;
  _Float16*  wtB  = (_Float16*)w;

  dim3 blk256(256), blk128(128);
  dim3 gProjDD(Dc / 128, Mrows / 64);        // (8, 128)
  dim3 gFfn1(Fc / 128, FFN_CHUNK / 64);      // (32, 32)
  dim3 gFfn2(Dc / 128, FFN_CHUNK / 64);      // (8, 32)
  dim3 gTrDD(Dc / 32, Dc / 32);
  dim3 gTrDF(Fc / 32, Dc / 32);
  dim3 gTrFD(Dc / 32, Fc / 32);

  embed_kernel<<<Mrows, blk256, 0, stream>>>(tokens, emb, x);

  for (int l = 0; l < Lc; ++l) {
    const float* Wq_l = Wq + (size_t)l * Dc * Dc;
    const float* Wk_l = Wk + (size_t)l * Dc * Dc;
    const float* Wv_l = Wv + (size_t)l * Dc * Dc;
    const float* Wo_l = Wo + (size_t)l * Dc * Dc;
    const float* W1_l = W1 + (size_t)l * Dc * Fc;
    const float* W2_l = W2 + (size_t)l * Fc * Dc;

    ln_kernel<<<Mrows, blk256, 0, stream>>>(x, ln1_g + (size_t)l * Dc,
                                            ln1_b + (size_t)l * Dc,
                                            nullptr, h16);
    wtrans_kernel<<<gTrDD, blk256, 0, stream>>>(Wq_l, wtA, Dc, Dc);
    gemm_kernel<<<gProjDD, blk256, 0, stream>>>(h16, wtA, bq + (size_t)l * Dc,
        nullptr, nullptr, q16, Mrows, Dc, Dc, 0);
    wtrans_kernel<<<gTrDD, blk256, 0, stream>>>(Wk_l, wtA, Dc, Dc);
    gemm_kernel<<<gProjDD, blk256, 0, stream>>>(h16, wtA, bk + (size_t)l * Dc,
        nullptr, nullptr, k16, Mrows, Dc, Dc, 0);
    wtrans_kernel<<<gTrDD, blk256, 0, stream>>>(Wv_l, wtA, Dc, Dc);
    gemm_kernel<<<gProjDD, blk256, 0, stream>>>(h16, wtA, bv + (size_t)l * Dc,
        nullptr, nullptr, v16, Mrows, Dc, Dc, 0);

    attn_kernel<<<Bc * Hc * 8, blk128, 0, stream>>>(q16, k16, v16, h16);

    wtrans_kernel<<<gTrDD, blk256, 0, stream>>>(Wo_l, wtA, Dc, Dc);
    gemm_kernel<<<gProjDD, blk256, 0, stream>>>(h16, wtA, bo + (size_t)l * Dc,
        x, x, nullptr, Mrows, Dc, Dc, 0);

    ln_kernel<<<Mrows, blk256, 0, stream>>>(x, ln2_g + (size_t)l * Dc,
                                            ln2_b + (size_t)l * Dc,
                                            nullptr, h16);
    wtrans_kernel<<<gTrDF, blk256, 0, stream>>>(W1_l, wtA, Dc, Fc);
    wtrans_kernel<<<gTrFD, blk256, 0, stream>>>(W2_l, wtB, Fc, Dc);
    for (int c = 0; c < Mrows / FFN_CHUNK; ++c) {
      const _Float16* hc = h16 + (size_t)c * FFN_CHUNK * Dc;
      float*          xc = x   + (size_t)c * FFN_CHUNK * Dc;
      gemm_kernel<<<gFfn1, blk256, 0, stream>>>(hc, wtA, b1 + (size_t)l * Fc,
          nullptr, nullptr, f16b, FFN_CHUNK, Fc, Dc, 1);
      gemm_kernel<<<gFfn2, blk256, 0, stream>>>(f16b, wtB, b2 + (size_t)l * Dc,
          xc, xc, nullptr, FFN_CHUNK, Dc, Fc, 0);
    }
  }

  ln_kernel<<<Mrows, blk256, 0, stream>>>(x, lnf_g, lnf_b, (float*)d_out,
                                          nullptr);
}
